// GroupMamba_4999341932558
// MI455X (gfx1250) — matrix operand order repaired and checked by reading, compiled, not run on hardware
//
#include <hip/hip_runtime.h>
#include <hip/hip_bf16.h>

typedef _Float16 h16 __attribute__((ext_vector_type(16)));
typedef _Float16 h8  __attribute__((ext_vector_type(8)));
typedef _Float16 h4  __attribute__((ext_vector_type(4)));
typedef float    f8  __attribute__((ext_vector_type(8)));

#define BATCH 8
#define H96   96
#define W96   96
#define NSEQ  (H96 * W96)   /* 9216 */
#define LSEQ  NSEQ
#define CCH   256
#define DI    64

// ---------- helpers ----------
__device__ __forceinline__ float sigf(float v) { return 1.0f / (1.0f + __expf(-v)); }
__device__ __forceinline__ float siluf(float v) { return v * sigf(v); }

__device__ __forceinline__ int seq_to_spatial(int l, int g) {
  switch (g) {
    case 0: return l;
    case 1: return LSEQ - 1 - l;
    case 2: return (l % H96) * W96 + (l / H96);
    default: { int t = LSEQ - 1 - l; return (t % H96) * W96 + (t / H96); }
  }
}
__device__ __forceinline__ int spatial_to_seq(int h, int w, int g) {
  switch (g) {
    case 0: return h * W96 + w;
    case 1: return LSEQ - 1 - (h * W96 + w);
    case 2: return w * H96 + h;
    default: return LSEQ - 1 - (w * H96 + h);
  }
}

// ---------- weight f32 -> f16 transpose (dst[k*rows+n] = src[n*cols+k]) ----------
__global__ void k_trans16(const float* __restrict__ src, _Float16* __restrict__ dst,
                          int rows, int cols) {
  int i = blockIdx.x * 256 + threadIdx.x;
  if (i >= rows * cols) return;
  int n = i / cols, k = i % cols;
  dst[k * rows + n] = (_Float16)src[i];
}

// ---------- WMMA GEMM: C[M,Nn] = A[M,K](f16) x B[K,Nn](f16) + bias ----------
// 256 thr = 8 wave32; block tile 128(M) x 32(N); each wave: one A frag -> 2 WMMAs.
// A tile staged via async global->LDS (CDNA5 ASYNCcnt path).
// Requires M%128==0, Nn%32==0, K%32==0 (all shapes here satisfy this).
__global__ __launch_bounds__(256)
void k_gemm_f16(const _Float16* __restrict__ A, int lda,
                const _Float16* __restrict__ Bm, int ldb,
                const float* __restrict__ bias,
                void* __restrict__ Cout, int ldc, int K, int out_f16) {
  __shared__ _Float16 sA[128][32];
  __shared__ _Float16 sBT[32][32];          // transposed: [n][k]
  const int t = threadIdx.x;
  const int wave = t >> 5, lane = t & 31;
  const int mbase = blockIdx.x * 128;
  const int nbase = blockIdx.y * 32;
  f8 acc0 = {}, acc1 = {};
  // staging coords (fixed per thread)
  const int arow = t >> 1;
  const int aseg = (t & 1) << 4;            // 0 or 16 halfs (32B)
  const int bkk = t >> 3;                   // 0..31
  const int bnn = (t & 7) << 2;             // 0..28
  const unsigned ldsA0 = (unsigned)(unsigned long long)&sA[arow][aseg];
  const unsigned ldsA1 = ldsA0 + 16;
  for (int k0 = 0; k0 < K; k0 += 32) {
    // --- async stage A tile (128x32 f16): 2 x b128 per thread, VGPR-bypassing
    {
      unsigned long long ga = (unsigned long long)(A + (size_t)(mbase + arow) * lda + k0 + aseg);
      asm volatile("global_load_async_to_lds_b128 %0, %1, off"
                   :: "v"(ldsA0), "v"(ga) : "memory");
      asm volatile("global_load_async_to_lds_b128 %0, %1, off"
                   :: "v"(ldsA1), "v"(ga + 16ull) : "memory");
    }
    // --- stage B tile (32x32) transposed into sBT[n][k]
    {
      h4 q = *(const h4*)(Bm + (size_t)(k0 + bkk) * ldb + nbase + bnn);
      sBT[bnn + 0][bkk] = q[0];
      sBT[bnn + 1][bkk] = q[1];
      sBT[bnn + 2][bkk] = q[2];
      sBT[bnn + 3][bkk] = q[3];
    }
    if (k0 + 32 < K) // prefetch next A slab -> global_prefetch_b8
      __builtin_prefetch(A + (size_t)(mbase + arow) * lda + k0 + 32, 0, 1);
    asm volatile("s_wait_asynccnt 0x0" ::: "memory");
    __syncthreads();

    // A fragment (16-bit A 16x32): lanes0-15 K0..15, lanes16-31 K16..31 (matches B)
    int mrow = (wave << 4) + (lane & 15);
    int kb = (lane < 16) ? 0 : 16;
    h8 alo = *(h8*)&sA[mrow][kb];
    h8 ahi = *(h8*)&sA[mrow][kb + 8];
    h16 af;
#pragma unroll
    for (int j = 0; j < 8; ++j) { af[j] = alo[j]; af[j + 8] = ahi[j]; }
    // B fragments: lane n = lane&15 (+16 for second tile); lanes0-15 K0..15, lanes16-31 K16..31
    int ncol = lane & 15;
    int kb2 = (lane < 16) ? 0 : 16;
    h16 bf0 = *(h16*)&sBT[ncol][kb2];
    h16 bf1 = *(h16*)&sBT[ncol + 16][kb2];

    acc0 = __builtin_amdgcn_wmma_f32_16x16x32_f16(false, af, false, bf0,
                                                  (short)0, acc0, false, false);
    acc1 = __builtin_amdgcn_wmma_f32_16x16x32_f16(false, af, false, bf1,
                                                  (short)0, acc1, false, false);
    __syncthreads();
  }
  // C/D layout: VGPR r -> M = r (lanes0-15) or 8+r (lanes16-31); N = lane&15
  int col = nbase + (lane & 15);
  float b0 = bias ? bias[col] : 0.0f;
  float b1 = bias ? bias[col + 16] : 0.0f;
  int rbase = mbase + (wave << 4) + ((lane < 16) ? 0 : 8);
  if (out_f16) {
    _Float16* C16 = (_Float16*)Cout;
#pragma unroll
    for (int r = 0; r < 8; ++r) {
      C16[(size_t)(rbase + r) * ldc + col]      = (_Float16)(acc0[r] + b0);
      C16[(size_t)(rbase + r) * ldc + col + 16] = (_Float16)(acc1[r] + b1);
    }
  } else {
    float* C32 = (float*)Cout;
#pragma unroll
    for (int r = 0; r < 8; ++r) {
      C32[(size_t)(rbase + r) * ldc + col]      = acc0[r] + b0;
      C32[(size_t)(rbase + r) * ldc + col + 16] = acc1[r] + b1;
    }
  }
}

// ---------- LayerNorm over C=256 -> f16 ----------
__global__ __launch_bounds__(256)
void k_ln_row(const float* __restrict__ x, const float* __restrict__ w,
              const float* __restrict__ b, _Float16* __restrict__ o16) {
  __shared__ float red[256];
  size_t row = blockIdx.x;
  int t = threadIdx.x;
  float v = x[row * CCH + t];
  red[t] = v; __syncthreads();
  for (int s = 128; s > 0; s >>= 1) { if (t < s) red[t] += red[t + s]; __syncthreads(); }
  float mean = red[0] / CCH; __syncthreads();
  float d = v - mean;
  red[t] = d * d; __syncthreads();
  for (int s = 128; s > 0; s >>= 1) { if (t < s) red[t] += red[t + s]; __syncthreads(); }
  o16[row * CCH + t] = (_Float16)(d * rsqrtf(red[0] / CCH + 1e-5f) * w[t] + b[t]);
}

// ---------- z_avg = mean over N (reads f16 LN output) ----------
__global__ __launch_bounds__(256)
void k_zavg(const _Float16* __restrict__ xn, float* __restrict__ zavg) {
  int b = blockIdx.x, c = threadIdx.x;
  const _Float16* p = xn + (size_t)b * NSEQ * CCH + c;
  float s = 0.0f;
  for (int n = 0; n < NSEQ; ++n) s += (float)p[(size_t)n * CCH];
  zavg[b * CCH + c] = s / (float)NSEQ;
}

// ---------- gate = sigmoid(zavg@fc_w.T + fc_b + conv1d(zavg)) ----------
__global__ __launch_bounds__(256)
void k_gate(const float* __restrict__ zavg, const float* __restrict__ fcw,
            const float* __restrict__ fcb, const float* __restrict__ c1w,
            float* __restrict__ gate) {
  int b = blockIdx.x, c = threadIdx.x;
  const float* z = zavg + b * CCH;
  float acc = fcb[c];
  for (int k = 0; k < CCH; ++k) acc += z[k] * fcw[c * CCH + k];
#pragma unroll
  for (int tk = 0; tk < 3; ++tk) {
    int idx = c - 1 + tk;
    if (idx >= 0 && idx < CCH) acc += c1w[tk] * z[idx];
  }
  gate[b * CCH + c] = sigf(acc);
}

// ---------- depthwise 3x3 conv + bias + SiLU : xz16[:, :64] -> xconv[b][d][hw] ----------
__global__ __launch_bounds__(64)
void k_dwconv(const _Float16* __restrict__ xz, const float* __restrict__ w2,
              const float* __restrict__ cb, float* __restrict__ xconv) {
  int pix = blockIdx.x;
  int d = threadIdx.x;
  int b = pix / NSEQ, hw = pix % NSEQ;
  int h = hw / W96, wq = hw % W96;
  float acc = cb[d];
#pragma unroll
  for (int ky = 0; ky < 3; ++ky) {
    int hh = h + ky - 1;
    if (hh < 0 || hh >= H96) continue;
#pragma unroll
    for (int kx = 0; kx < 3; ++kx) {
      int ww = wq + kx - 1;
      if (ww < 0 || ww >= W96) continue;
      acc += w2[d * 9 + ky * 3 + kx] *
             (float)xz[((size_t)b * NSEQ + hh * W96 + ww) * 128 + d];
    }
  }
  xconv[((size_t)(b * DI + d)) * NSEQ + hw] = siluf(acc);
}

// ---------- x_proj dots: 6 planes per (b,l): dtr[0..3], B, C  (only 1.8MB!) ----------
__global__ __launch_bounds__(256)
void k_dots(const float* __restrict__ xconv, const float* __restrict__ xpw,
            float* __restrict__ dots, int g) {
  size_t gid = (size_t)blockIdx.x * 256 + threadIdx.x;  // b*L + l
  int b = (int)(gid / LSEQ), l = (int)(gid % LSEQ);
  int sidx = seq_to_spatial(l, g);
  const float* xc = xconv + (size_t)b * DI * NSEQ + sidx;
  float dot[6] = {0, 0, 0, 0, 0, 0};
  for (int d = 0; d < DI; ++d) {
    float s = xc[(size_t)d * NSEQ];
#pragma unroll
    for (int j = 0; j < 6; ++j) dot[j] += s * xpw[j * DI + d];
  }
#pragma unroll
  for (int j = 0; j < 6; ++j)
    dots[((size_t)b * 6 + j) * LSEQ + l] = dot[j];
}

// ---------- chunked parallel scan; dt/dA/dBx recomputed on the fly (no HBM round-trip) ----------
__global__ __launch_bounds__(256)
void k_scan(const float* __restrict__ xconv, const float* __restrict__ dots,
            const float* __restrict__ dtw, const float* __restrict__ dtb,
            const float* __restrict__ A_log, float* __restrict__ ys, int g) {
  __shared__ float sAg[256], sBg[256];
  int bd = blockIdx.x;                  // b*64 + d
  int b = bd >> 6, d = bd & 63;
  int t = threadIdx.x;
  const int CL = LSEQ / 256;            // 36
  float w0 = dtw[d * 4 + 0], w1 = dtw[d * 4 + 1];
  float w2 = dtw[d * 4 + 2], w3 = dtw[d * 4 + 3];
  float db = dtb[d];
  float Aval = -__expf(A_log[d]);
  const float* dp = dots + (size_t)b * 6 * LSEQ;
  const float* xcp = xconv + (size_t)(b * DI + d) * NSEQ;
  int l0 = t * CL;
  float aa = 1.0f, bb = 0.0f;
  for (int i = 0; i < CL; ++i) {        // pass 1: chunk aggregate
    int l = l0 + i;
    float dtv = db + dp[l] * w0 + dp[LSEQ + l] * w1 + dp[2 * LSEQ + l] * w2 + dp[3 * LSEQ + l] * w3;
    float dts = (dtv > 20.0f) ? dtv : log1pf(__expf(dtv));
    float a = __expf(dts * Aval);
    float bx = dts * xcp[seq_to_spatial(l, g)] * dp[4 * LSEQ + l];
    bb = a * bb + bx;
    aa *= a;
  }
  sAg[t] = aa; sBg[t] = bb;
  __syncthreads();
  for (int off = 1; off < 256; off <<= 1) {   // inclusive scan of (a,b) pairs
    float pa = 0.0f, pb = 0.0f, ca = sAg[t], cb2 = sBg[t];
    bool has = (t >= off);
    if (has) { pa = sAg[t - off]; pb = sBg[t - off]; }
    __syncthreads();
    if (has) { sAg[t] = ca * pa; sBg[t] = ca * pb + cb2; }
    __syncthreads();
  }
  float h = (t == 0) ? 0.0f : sBg[t - 1];
  for (int i = 0; i < CL; ++i) {        // pass 2: replay + y = h*C (layout [b][l][d])
    int l = l0 + i;
    float dtv = db + dp[l] * w0 + dp[LSEQ + l] * w1 + dp[2 * LSEQ + l] * w2 + dp[3 * LSEQ + l] * w3;
    float dts = (dtv > 20.0f) ? dtv : log1pf(__expf(dtv));
    float a = __expf(dts * Aval);
    float bx = dts * xcp[seq_to_spatial(l, g)] * dp[4 * LSEQ + l];
    h = a * h + bx;
    ys[((size_t)b * LSEQ + l) * DI + d] = h * dp[5 * LSEQ + l];
  }
}

// ---------- from_seq + Ds*seq + LN(Di) + silu(z) gate -> f16 activation ----------
__global__ __launch_bounds__(64)
void k_post(const float* __restrict__ ys, const float* __restrict__ xconv,
            const _Float16* __restrict__ xz, const float* __restrict__ Dsv,
            const float* __restrict__ onw, const float* __restrict__ onb,
            _Float16* __restrict__ act, int g) {
  __shared__ float red[64];
  int pix = blockIdx.x;
  int d = threadIdx.x;
  int b = pix / NSEQ, hw = pix % NSEQ;
  int h = hw / W96, wq = hw % W96;
  int l = spatial_to_seq(h, wq, g);
  float y = ys[((size_t)b * LSEQ + l) * DI + d] +          // coalesced over d
            Dsv[d] * xconv[((size_t)(b * DI + d)) * NSEQ + hw];
  red[d] = y; __syncthreads();
  for (int s = 32; s > 0; s >>= 1) { if (d < s) red[d] += red[d + s]; __syncthreads(); }
  float mean = red[0] / (float)DI; __syncthreads();
  float dv = y - mean;
  red[d] = dv * dv; __syncthreads();
  for (int s = 32; s > 0; s >>= 1) { if (d < s) red[d] += red[d + s]; __syncthreads(); }
  float yn = dv * rsqrtf(red[0] / (float)DI + 1e-5f) * onw[d] + onb[d];
  float z = (float)xz[(size_t)pix * 128 + DI + d];
  act[(size_t)pix * DI + d] = (_Float16)(yn * siluf(z));
}

// ---------- xm = xm*skip*xn*gate ; LN(C) -> f16 ----------
__global__ __launch_bounds__(256)
void k_combine_ln(const float* __restrict__ xm, const _Float16* __restrict__ xn,
                  const float* __restrict__ gate, const float* __restrict__ skip,
                  const float* __restrict__ nw, const float* __restrict__ nb,
                  _Float16* __restrict__ o16) {
  __shared__ float red[256];
  size_t row = blockIdx.x;
  int t = threadIdx.x;
  int b = (int)(row / NSEQ);
  float v = xm[row * CCH + t] * skip[0] * (float)xn[row * CCH + t] * gate[b * CCH + t];
  red[t] = v; __syncthreads();
  for (int s = 128; s > 0; s >>= 1) { if (t < s) red[t] += red[t + s]; __syncthreads(); }
  float mean = red[0] / CCH; __syncthreads();
  float d = v - mean;
  red[t] = d * d; __syncthreads();
  for (int s = 128; s > 0; s >>= 1) { if (t < s) red[t] += red[t + s]; __syncthreads(); }
  o16[row * CCH + t] = (_Float16)(d * rsqrtf(red[0] / CCH + 1e-5f) * nw[t] + nb[t]);
}

// ---------- host ----------
extern "C" void kernel_launch(void* const* d_in, const int* in_sizes, int n_in,
                              void* d_out, int out_size, void* d_ws, size_t ws_size,
                              hipStream_t stream) {
  (void)in_sizes; (void)n_in; (void)out_size; (void)ws_size;
  const float* x          = (const float*)d_in[0];
  const float* norm_w     = (const float*)d_in[1];
  const float* norm_b     = (const float*)d_in[2];
  const float* fc_w       = (const float*)d_in[3];
  const float* fc_b       = (const float*)d_in[4];
  const float* conv1d_w   = (const float*)d_in[5];
  const float* in_proj_w  = (const float*)d_in[6];
  const float* in_proj_b  = (const float*)d_in[7];
  const float* conv2d_w   = (const float*)d_in[8];
  const float* conv2d_b   = (const float*)d_in[9];
  const float* x_proj_w   = (const float*)d_in[10];
  const float* dt_w       = (const float*)d_in[11];
  const float* dt_b       = (const float*)d_in[12];
  const float* A_log      = (const float*)d_in[13];
  const float* Ds         = (const float*)d_in[14];
  const float* out_norm_w = (const float*)d_in[15];
  const float* out_norm_b = (const float*)d_in[16];
  const float* out_proj_w = (const float*)d_in[17];
  const float* out_proj_b = (const float*)d_in[18];
  const float* proj_w     = (const float*)d_in[19];
  const float* proj_b     = (const float*)d_in[20];
  const float* skip       = (const float*)d_in[21];
  float* out = (float*)d_out;

  char* ws = (char*)d_ws;
  size_t off = 0;
  auto take = [&](size_t bytes) -> char* {
    char* p = ws + off;
    off += (bytes + 255) & ~(size_t)255;
    return p;
  };
  const size_t BN = (size_t)BATCH * NSEQ;   // 73728 rows
  _Float16* xn16   = (_Float16*)take(BN * CCH * 2);
  _Float16* xzh    = (_Float16*)take(BN * 128 * 2);
  float*    xconv  = (float*)take(BN * DI * 4);
  float*    dots   = (float*)take((size_t)BATCH * 6 * LSEQ * 4);
  float*    ysb    = (float*)take(BN * DI * 4);
  _Float16* act16  = (_Float16*)take(BN * DI * 2);
  float*    xm     = (float*)take(BN * CCH * 4);
  _Float16* xmln16 = (_Float16*)take(BN * CCH * 2);
  float*    zavg   = (float*)take(BATCH * CCH * 4);
  float*    gate   = (float*)take(BATCH * CCH * 4);
  _Float16* wip16  = (_Float16*)take(4 * 128 * 64 * 2);
  _Float16* wop16  = (_Float16*)take(4 * 64 * 64 * 2);
  _Float16* wpj16  = (_Float16*)take(256 * 256 * 2);

  // weights -> f16 (transposed to KxN)
  for (int g = 0; g < 4; ++g) {
    k_trans16<<<(128 * 64 + 255) / 256, 256, 0, stream>>>(in_proj_w + g * 128 * 64,
                                                          wip16 + g * 128 * 64, 128, 64);
    k_trans16<<<(64 * 64 + 255) / 256, 256, 0, stream>>>(out_proj_w + g * 64 * 64,
                                                         wop16 + g * 64 * 64, 64, 64);
  }
  k_trans16<<<(256 * 256 + 255) / 256, 256, 0, stream>>>(proj_w, wpj16, 256, 256);

  k_ln_row<<<(unsigned)BN, 256, 0, stream>>>(x, norm_w, norm_b, xn16);
  k_zavg<<<BATCH, 256, 0, stream>>>(xn16, zavg);
  k_gate<<<BATCH, 256, 0, stream>>>(zavg, fc_w, fc_b, conv1d_w, gate);

  for (int g = 0; g < 4; ++g) {
    dim3 gin((unsigned)(BN / 128), 128 / 32);
    k_gemm_f16<<<gin, 256, 0, stream>>>(xn16 + g * 64, CCH, wip16 + g * 128 * 64, 128,
                                        in_proj_b + g * 128, xzh, 128, 64, 1);
    k_dwconv<<<(unsigned)BN, 64, 0, stream>>>(xzh, conv2d_w + g * DI * 9, conv2d_b + g * DI, xconv);
    k_dots<<<(unsigned)(BN / 256), 256, 0, stream>>>(xconv, x_proj_w + g * 6 * DI, dots, g);
    k_scan<<<BATCH * DI, 256, 0, stream>>>(xconv, dots, dt_w + g * DI * 4, dt_b + g * DI,
                                           A_log + g * DI, ysb, g);
    k_post<<<(unsigned)BN, 64, 0, stream>>>(ysb, xconv, xzh, Ds + g * DI,
                                            out_norm_w + g * DI, out_norm_b + g * DI, act16, g);
    dim3 gop((unsigned)(BN / 128), 64 / 32);
    k_gemm_f16<<<gop, 256, 0, stream>>>(act16, DI, wop16 + g * 64 * 64, 64,
                                        out_proj_b + g * 64, xm + g * 64, CCH, 64, 0);
  }

  k_combine_ln<<<(unsigned)BN, 256, 0, stream>>>(xm, xn16, gate, skip, norm_w, norm_b, xmln16);
  dim3 gpr((unsigned)(BN / 128), 256 / 32);
  k_gemm_f16<<<gpr, 256, 0, stream>>>(xmln16, CCH, wpj16, 256, proj_b, out, CCH, 256, 0);
}
